// ShiftWindowMSA_48747878810064
// MI455X (gfx1250) — matrix operand order repaired and checked by reading, not compiled, not run on hardware
//
#include <hip/hip_runtime.h>

typedef __attribute__((ext_vector_type(16))) __bf16 v16bf;
typedef __attribute__((ext_vector_type(8)))  float  v8f;
typedef unsigned short u16;
typedef unsigned int   u32;

#define WSZ    8
#define SHIFTA 4
#define NHEADS 12
#define CDIM   384
#define HD     32
#define NTOK   64
#define HW     96
#define NWX    12
#define NWIN   144
#define NBATCH 16
#define QSCALE 0.17677669529663687f

#define XST 392   // row stride (halfs) for 64x384 token-major tiles (pad 8 -> 4-bank skew)
#define VTT 72    // row stride (halfs) for transposed V (384 x 64)
#define PST 72    // row stride (halfs) for per-wave P scratch (32 x 64)

union F8 { v8f v; float e[8]; };

static __device__ __forceinline__ u16 f2bf(float f) {
  u32 u = __float_as_uint(f);
  u32 r = u + 0x7FFFu + ((u >> 16) & 1u);   // round-to-nearest-even bf16
  return (u16)(r >> 16);
}
static __device__ __forceinline__ u32 pk2(float lo, float hi) {
  return (u32)f2bf(lo) | ((u32)f2bf(hi) << 16);
}

// A fragment (16x32 bf16). lane L: row = rowp's row, elems 0..7 -> k=kbase+hi*8+0..7,
// elems 8..15 -> k=kbase+16+hi*8+0..7  (ISA 7.12.2 16-bit A layout)
static __device__ __forceinline__ v16bf load_a(const u16* rowp, int kbase, int hi) {
  union { v16bf v; uint4 q[2]; } u;
  u.q[0] = *(const uint4*)(rowp + kbase + hi * 8);
  u.q[1] = *(const uint4*)(rowp + kbase + 16 + hi * 8);
  return u.v;
}
// B fragment (32x16 bf16). lane L: col = L&15, elems 0..7 -> k=kbase+hi*8+0..7,
// elems 8..15 -> k=kbase+16+hi*8+0..7
static __device__ __forceinline__ v16bf load_b(const u16* rowp, int kbase, int hi) {
  union { v16bf v; uint4 q[2]; } u;
  u.q[0] = *(const uint4*)(rowp + kbase + hi * 8);
  u.q[1] = *(const uint4*)(rowp + kbase + 16 + hi * 8);
  return u.v;
}
static __device__ __forceinline__ v8f wmma_bf16(v16bf a, v16bf b, v8f c) {
  return __builtin_amdgcn_wmma_f32_16x16x32_bf16(false, a, false, b, (short)0, c, false, false);
}

// -------- weight prep: f32 -> bf16 + transpose (B operand wants K contiguous) --------
__global__ __launch_bounds__(256)
void prep_weights(const float* __restrict__ Wqkv, const float* __restrict__ Wskip,
                  const float* __restrict__ Wproj,
                  u16* __restrict__ wqkvT, u16* __restrict__ wskipT, u16* __restrict__ wprojT) {
  int i = blockIdx.x * 256 + threadIdx.x;
  if (i < 768 * 384) {
    int n = i / 384, k = i - n * 384;
    wqkvT[i] = f2bf(Wqkv[k * 768 + n]);
  } else if (i < 768 * 384 + 384 * 384) {
    int j = i - 768 * 384;
    int n = j / 384, k = j - n * 384;
    wskipT[j] = f2bf(Wskip[k * 384 + n]);
  } else if (i < 768 * 384 + 2 * 384 * 384) {
    int j = i - (768 * 384 + 384 * 384);
    int n = j / 384, k = j - n * 384;
    wprojT[j] = f2bf(Wproj[k * 384 + n]);
  }
}

// -------- fused shifted-window MSA: one workgroup (8 wave32) per window --------
__global__ __launch_bounds__(256)
void swin_msa_kernel(const float* __restrict__ query, const float* __restrict__ skipq,
                     const u16* __restrict__ WqkvT, const float* __restrict__ bqkv,
                     const u16* __restrict__ WskipT, const float* __restrict__ bskip,
                     const float* __restrict__ rpb,
                     const u16* __restrict__ WprojT, const float* __restrict__ bproj,
                     float* __restrict__ out) {
  __shared__ __align__(16) u16 Xs [NTOK * XST];   // X tile -> later S tile -> later O tile
  __shared__ __align__(16) u16 Ksh[NTOK * XST];   // K, token-major
  __shared__ __align__(16) u16 Vt [CDIM * VTT];   // V, transposed (d-major)
  __shared__ __align__(16) u16 Qs [NTOK * XST];   // Q (pre-scaled), token-major
  __shared__ __align__(16) u16 Ps [8 * 32 * PST]; // per-wave softmax probs (32 rows x 64)

  const int win  = blockIdx.x;
  const int bb   = win / NWIN;
  const int wrem = win - bb * NWIN;
  const int wh   = wrem / NWX;
  const int ww   = wrem - wh * NWX;

  const int tid  = threadIdx.x;
  const int lane = tid & 31;
  const int wv   = tid >> 5;
  const int hi   = lane >> 4;
  const int n15  = lane & 15;

  const size_t img_base = (size_t)bb * (HW * HW);

  // ---- phase 1: gather rolled X window -> Xs (bf16) ----
  {
    for (int i = tid; i < NTOK * (CDIM / 4); i += 256) {
      int t  = i / (CDIM / 4);
      int c4 = (i - t * (CDIM / 4)) * 4;
      int r = t >> 3, cc = t & 7;
      int ho = wh * WSZ + r + SHIFTA;  if (ho >= HW) ho -= HW;
      int wo = ww * WSZ + cc + SHIFTA; if (wo >= HW) wo -= HW;
      const float4 v = *(const float4*)(query + (img_base + (size_t)(ho * HW + wo)) * CDIM + c4);
      *(uint2*)&Xs[t * XST + c4] = make_uint2(pk2(v.x, v.y), pk2(v.z, v.w));
    }
  }
  __syncthreads();

  // ---- phase 2: KV = X @ Wqkv + b ;  K -> Ksh (token-major), V -> Vt (transposed) ----
  for (int tile = wv; tile < 4 * 48; tile += 8) {
    const int mt = tile & 3;
    const int nt = tile >> 2;
    const u16* arow = Xs + (mt * 16 + n15) * XST;
    const u16* brow = WqkvT + (size_t)(nt * 16 + n15) * CDIM;
    v8f acc = {0.f,0.f,0.f,0.f,0.f,0.f,0.f,0.f};
#pragma unroll
    for (int kk = 0; kk < CDIM / 32; ++kk)
      acc = wmma_bf16(load_a(arow, kk * 32, hi), load_b(brow, kk * 32, hi), acc);
    const int col = nt * 16 + n15;
    const float bias = bqkv[col];
    F8 c; c.v = acc;
    if (nt < 24) {                       // K columns 0..383
#pragma unroll
      for (int j = 0; j < 8; ++j)
        Ksh[(mt * 16 + hi * 8 + j) * XST + col] = f2bf(c.e[j] + bias);
    } else {                             // V columns 384..767, store transposed
      const int d = col - CDIM;
      uint4 p;
      p.x = pk2(c.e[0] + bias, c.e[1] + bias);
      p.y = pk2(c.e[2] + bias, c.e[3] + bias);
      p.z = pk2(c.e[4] + bias, c.e[5] + bias);
      p.w = pk2(c.e[6] + bias, c.e[7] + bias);
      *(uint4*)&Vt[d * VTT + mt * 16 + hi * 8] = p;
    }
  }
  __syncthreads();

  // ---- phase 3: gather rolled S window -> Xs (X is dead) ----
  {
    for (int i = tid; i < NTOK * (CDIM / 4); i += 256) {
      int t  = i / (CDIM / 4);
      int c4 = (i - t * (CDIM / 4)) * 4;
      int r = t >> 3, cc = t & 7;
      int ho = wh * WSZ + r + SHIFTA;  if (ho >= HW) ho -= HW;
      int wo = ww * WSZ + cc + SHIFTA; if (wo >= HW) wo -= HW;
      const float4 v = *(const float4*)(skipq + (img_base + (size_t)(ho * HW + wo)) * CDIM + c4);
      *(uint2*)&Xs[t * XST + c4] = make_uint2(pk2(v.x, v.y), pk2(v.z, v.w));
    }
  }
  __syncthreads();

  // ---- phase 4: Q = (S @ Wskip + b) * SCALE -> Qs ----
  for (int tile = wv; tile < 4 * 24; tile += 8) {
    const int mt = tile & 3;
    const int nt = tile >> 2;
    const u16* arow = Xs + (mt * 16 + n15) * XST;
    const u16* brow = WskipT + (size_t)(nt * 16 + n15) * CDIM;
    v8f acc = {0.f,0.f,0.f,0.f,0.f,0.f,0.f,0.f};
#pragma unroll
    for (int kk = 0; kk < CDIM / 32; ++kk)
      acc = wmma_bf16(load_a(arow, kk * 32, hi), load_b(brow, kk * 32, hi), acc);
    const int col = nt * 16 + n15;
    const float bias = bskip[col];
    F8 c; c.v = acc;
#pragma unroll
    for (int j = 0; j < 8; ++j)
      Qs[(mt * 16 + hi * 8 + j) * XST + col] = f2bf((c.e[j] + bias) * QSCALE);
  }
  __syncthreads();

  // ---- phase 5: attention. 4 heads at a time, 2 waves per head (32 rows each) ----
  const bool edgeH = (wh == NWX - 1);
  const bool edgeW = (ww == NWX - 1);
  const bool edge  = edgeH || edgeW;
  for (int hb = 0; hb < NHEADS; hb += 4) {
    const int h     = hb + (wv >> 1);
    const int mhalf = wv & 1;

    // S = Q_h @ K_h^T  (2 row-tiles x 4 col-tiles, K=32 -> one WMMA each)
    v16bf afr[2];
#pragma unroll
    for (int mi = 0; mi < 2; ++mi)
      afr[mi] = load_a(Qs + ((mhalf * 2 + mi) * 16 + n15) * XST, h * HD, hi);
    v16bf bfr[4];
#pragma unroll
    for (int nt = 0; nt < 4; ++nt)
      bfr[nt] = load_b(Ksh + (nt * 16 + n15) * XST, h * HD, hi);
    F8 s[2][4];
#pragma unroll
    for (int mi = 0; mi < 2; ++mi)
#pragma unroll
      for (int nt = 0; nt < 4; ++nt) {
        v8f z = {0.f,0.f,0.f,0.f,0.f,0.f,0.f,0.f};
        s[mi][nt].v = wmma_bf16(afr[mi], bfr[nt], z);
      }

    // + rel-pos bias + shift mask, row softmax (rows live in one 16-lane half)
#pragma unroll
    for (int mi = 0; mi < 2; ++mi) {
      const int mrow0 = (mhalf * 2 + mi) * 16 + hi * 8;
#pragma unroll
      for (int j = 0; j < 8; ++j) {
        const int arow = mrow0 + j;
        const int ra = 15 * (arow >> 3) + (arow & 7);
        int cnt_a = 0;
        if (edge) {
          int ia = arow >> 3, ja = arow & 7;
          int hida = edgeH ? (1 + (ia >= SHIFTA)) : 0;
          int wida = edgeW ? (1 + (ja >= SHIFTA)) : 0;
          cnt_a = hida * 3 + wida;
        }
        float pv[4];
        float mx = -3.4e38f;
#pragma unroll
        for (int nt = 0; nt < 4; ++nt) {
          const int bt = nt * 16 + n15;
          const int bf = 63 - bt;                       // np.flip along axis 1
          const int rb = 15 * (bf >> 3) + (bf & 7);
          float val = s[mi][nt].e[j] + rpb[(ra + rb) * NHEADS + h];
          if (edge) {
            int ib = bt >> 3, jb = bt & 7;
            int hidb = edgeH ? (1 + (ib >= SHIFTA)) : 0;
            int widb = edgeW ? (1 + (jb >= SHIFTA)) : 0;
            if (hidb * 3 + widb != cnt_a) val -= 100.0f;
          }
          pv[nt] = val;
          mx = fmaxf(mx, val);
        }
        mx = fmaxf(mx, __shfl_xor(mx, 1, 32));
        mx = fmaxf(mx, __shfl_xor(mx, 2, 32));
        mx = fmaxf(mx, __shfl_xor(mx, 4, 32));
        mx = fmaxf(mx, __shfl_xor(mx, 8, 32));
        float sum = 0.f;
#pragma unroll
        for (int nt = 0; nt < 4; ++nt) { pv[nt] = __expf(pv[nt] - mx); sum += pv[nt]; }
        sum += __shfl_xor(sum, 1, 32);
        sum += __shfl_xor(sum, 2, 32);
        sum += __shfl_xor(sum, 4, 32);
        sum += __shfl_xor(sum, 8, 32);
        const float inv = 1.0f / sum;
        const int lrow = mi * 16 + hi * 8 + j;
#pragma unroll
        for (int nt = 0; nt < 4; ++nt)
          Ps[(wv * 32 + lrow) * PST + nt * 16 + n15] = f2bf(pv[nt] * inv);
      }
    }

    // O_h = P @ V_h  (A from per-wave P scratch, B from transposed V -> contiguous b128)
    const u16* pbase = Ps + wv * 32 * PST;
#pragma unroll
    for (int mi = 0; mi < 2; ++mi) {
      const u16* prow = pbase + (mi * 16 + n15) * PST;
#pragma unroll
      for (int dt = 0; dt < 2; ++dt) {
        const int d = h * HD + dt * 16 + n15;
        v8f acc = {0.f,0.f,0.f,0.f,0.f,0.f,0.f,0.f};
#pragma unroll
        for (int ks = 0; ks < 2; ++ks)
          acc = wmma_bf16(load_a(prow, ks * 32, hi), load_b(Vt + d * VTT, ks * 32, hi), acc);
        F8 o; o.v = acc;
#pragma unroll
        for (int j = 0; j < 8; ++j) {
          int t = (mhalf * 2 + mi) * 16 + hi * 8 + j;
          Xs[t * XST + d] = f2bf(o.e[j]);   // O reuses Xs (S is dead)
        }
      }
    }
  }
  __syncthreads();

  // ---- phase 6: out = O @ Wproj + b, scatter with reverse roll ----
  for (int tile = wv; tile < 4 * 24; tile += 8) {
    const int mt = tile & 3;
    const int nt = tile >> 2;
    const u16* arow = Xs + (mt * 16 + n15) * XST;
    const u16* brow = WprojT + (size_t)(nt * 16 + n15) * CDIM;
    v8f acc = {0.f,0.f,0.f,0.f,0.f,0.f,0.f,0.f};
#pragma unroll
    for (int kk = 0; kk < CDIM / 32; ++kk)
      acc = wmma_bf16(load_a(arow, kk * 32, hi), load_b(brow, kk * 32, hi), acc);
    const int col = nt * 16 + n15;
    const float bias = bproj[col];
    F8 c; c.v = acc;
#pragma unroll
    for (int j = 0; j < 8; ++j) {
      int t = mt * 16 + hi * 8 + j;
      int r = t >> 3, cc = t & 7;
      int ho = wh * WSZ + r + SHIFTA;  if (ho >= HW) ho -= HW;
      int wo = ww * WSZ + cc + SHIFTA; if (wo >= HW) wo -= HW;
      out[(img_base + (size_t)(ho * HW + wo)) * CDIM + col] = c.e[j] + bias;
    }
  }
}

extern "C" void kernel_launch(void* const* d_in, const int* in_sizes, int n_in,
                              void* d_out, int out_size, void* d_ws, size_t ws_size,
                              hipStream_t stream) {
  (void)in_sizes; (void)n_in; (void)out_size; (void)ws_size;
  const float* query = (const float*)d_in[0];
  const float* skipq = (const float*)d_in[1];
  const float* Wqkv  = (const float*)d_in[2];
  const float* bqkv  = (const float*)d_in[3];
  const float* Wskip = (const float*)d_in[4];
  const float* bskip = (const float*)d_in[5];
  const float* rpb   = (const float*)d_in[6];
  const float* Wproj = (const float*)d_in[7];
  const float* bproj = (const float*)d_in[8];
  // d_in[9], d_in[10] are h=96, w=96 (hardcoded)
  float* out = (float*)d_out;

  u16* wqkvT  = (u16*)d_ws;              // 768*384 bf16
  u16* wskipT = wqkvT  + 768 * 384;      // 384*384 bf16
  u16* wprojT = wskipT + 384 * 384;      // 384*384 bf16

  prep_weights<<<2304, 256, 0, stream>>>(Wqkv, Wskip, Wproj, wqkvT, wskipT, wprojT);
  swin_msa_kernel<<<NBATCH * NWIN, 256, 0, stream>>>(query, skipq, wqkvT, bqkv,
                                                     wskipT, bskip, rpb, wprojT, bproj, out);
}
